// RNN_86612310491990
// MI455X (gfx1250) — matrix operand order repaired and checked by reading, compiled, not run on hardware
//
#include <hip/hip_runtime.h>
#include <hip/hip_bf16.h>
#include <stdint.h>

// ---------------- problem constants ----------------
constexpr int   Tn = 150, Bn = 128, In = 64, Hn = 512;
constexpr float DT_C = 10.0f, TAU_MIN_C = 20.0f, TAU_MAX_C = 100.0f;
constexpr size_t BH  = (size_t)Bn * Hn;          // 65536
constexpr size_t TBH = (size_t)Tn * BH;          // 9830400
constexpr size_t HH  = (size_t)Hn * Hn;          // 262144

// ---------------- workspace layout (bytes) ----------------
constexpr size_t WOFF_WEFF = 0;                               // 7 * HH bf16 effective weights
constexpr size_t WOFF_WIN  = WOFF_WEFF + 7 * HH * 2;          // In*Hn bf16 w_in
constexpr size_t WOFF_STIM = WOFF_WIN  + (size_t)In * Hn * 2; // Tn*Bn*In bf16 stims
constexpr size_t WOFF_A    = WOFF_STIM + (size_t)Tn * Bn * In * 2; // 3*Hn f32 (DT/tau)
constexpr size_t WOFF_X    = WOFF_A    + 3 * Hn * 4;          // 3*BH f32 membrane state
constexpr size_t WOFF_R    = WOFF_X    + 3 * BH * 4;          // 2*3*BH bf16 rate ping-pong

// Weff matrix order: 0=Wr1 1=Wr2 2=Wr3 3=Wff12 4=Wff23 5=Wfb21 6=Wfb32
// out sections (each TBH floats): 0=h1 1=h2 2=h3 3=ff12 4=ff23 5=fb21 6=fb32, then taus (3*Hn)

typedef __attribute__((ext_vector_type(16))) __bf16 v16bf;
typedef __attribute__((ext_vector_type(8)))  float  v8f;

// ---------------- WMMA K-chain, B sourced from LDS column block ----------------
// aRowPtr: &act[row][0] for this lane's A row (lane%16 selects M).
// sB     : LDS column block, sB[k*2 + half] = W[k][n0 + half*8 .. +7] (uint4 = 8 bf16)
template <int K>
__device__ __forceinline__ v8f wmma_chain_lds(const __bf16* __restrict__ aRowPtr,
                                              const uint4* sB, int lane, int kb,
                                              v8f acc) {
#pragma unroll
  for (int k0 = 0; k0 < K; k0 += 32) {
    union { v16bf v; uint4 u[2]; } Am, Bm;
    // A 16x32 bf16: lanes 0-15 -> K = k0+{0..7, 16..23}; lanes 16-31 -> +8 (kb)
    Am.u[0] = *(const uint4*)(aRowPtr + k0 + kb);
    Am.u[1] = *(const uint4*)(aRowPtr + k0 + kb + 16);
    // B 32x16 bf16: lane%16 = N; K in the same order as A (gathered from LDS rows)
    const __bf16* sBh = (const __bf16*)sB;
#pragma unroll
    for (int s = 0; s < 8; s++) {
      Bm.v[s]     = sBh[(k0 + kb + s) * 16 + (lane & 15)];
      Bm.v[s + 8] = sBh[(k0 + kb + 16 + s) * 16 + (lane & 15)];
    }
    acc = __builtin_amdgcn_wmma_f32_16x16x32_bf16(false, Am.v, false, Bm.v,
                                                  (short)0, acc, false, false);
  }
  return acc;
}

// store one 16x16 f32 C tile (C layout: VGPR i -> M = i + 8*(lane/16), N = lane%16)
__device__ __forceinline__ void store_tile(float* __restrict__ dst, v8f v,
                                           int cm, int cn) {
#pragma unroll
  for (int i = 0; i < 8; i++) dst[(size_t)(cm + i) * Hn + cn] = v[i];
}

__device__ __forceinline__ void update_state(float* __restrict__ X,
                                             float* __restrict__ h,
                                             __bf16* __restrict__ rn,
                                             float a, v8f tot, int cm, int cn) {
#pragma unroll
  for (int i = 0; i < 8; i++) {
    size_t idx = (size_t)(cm + i) * Hn + cn;
    float nx = (1.0f - a) * X[idx] + a * tot[i];
    X[idx] = nx;
    float nr = fmaxf(nx, 0.0f);
    h[idx]  = nr;
    rn[idx] = (__bf16)nr;
  }
}

// ---------------- per-timestep kernel ----------------
// grid: 96 blocks x 256 threads. block = (layer, n-column-block); its 8 waves cover
// the 8 M(batch)-tiles, so the W[:, n0:n0+16] column blocks are shared via LDS.
__global__ __launch_bounds__(256) void rnn_step(
    const __bf16* __restrict__ Weff, const __bf16* __restrict__ winb,
    const __bf16* __restrict__ stimb, const float* __restrict__ a_ws,
    float* __restrict__ X, const __bf16* __restrict__ r_old,
    __bf16* __restrict__ r_new, float* __restrict__ out, int tstep) {
  __shared__ uint4 sW[3][Hn * 2];   // 3 chains x 512 rows x 32B = 48 KB
  __shared__ uint4 sWin[In * 2];    // w_in column block: 64 rows x 32B = 2 KB

  const int layer = blockIdx.x >> 5;         // 0..2
  const int n0    = (blockIdx.x & 31) << 4;  // hidden column-tile origin
  const int wave  = threadIdx.x >> 5;        // 0..7 -> batch tile
  const int lane  = threadIdx.x & 31;
  const int m0   = wave << 4;
  const int aRow = m0 + (lane & 15);         // A-matrix row for this lane
  const int kb   = (lane & 16) >> 1;         // 0 or 8: K sub-offset per lane half
  const int cm   = m0 + ((lane >> 4) << 3);  // C-tile row base
  const int cn   = n0 + (lane & 15);         // C-tile col

  // chains per layer: weight matrix index + source activation
  int mats[3] = {0, 0, 0};
  int nc;
  if      (layer == 0) { mats[0] = 0; mats[1] = 5;              nc = 2; }
  else if (layer == 1) { mats[0] = 3; mats[1] = 1; mats[2] = 6; nc = 3; }
  else                 { mats[0] = 4; mats[1] = 2;              nc = 2; }

  // cooperative stage of weight column blocks into LDS
  for (int c = 0; c < nc; c++) {
    const __bf16* W = Weff + (size_t)mats[c] * HH + n0;
    for (int j = threadIdx.x; j < Hn * 2; j += 256) {
      int k = j >> 1, half = j & 1;
      sW[c][j] = *(const uint4*)(W + (size_t)k * Hn + half * 8);
    }
  }
  if (layer == 0) {
    for (int j = threadIdx.x; j < In * 2; j += 256) {
      int k = j >> 1, half = j & 1;
      sWin[j] = *(const uint4*)(winb + (size_t)k * Hn + n0 + half * 8);
    }
  }
  __syncthreads();

  const __bf16* r1 = r_old;
  const __bf16* r2 = r_old + BH;
  const __bf16* r3 = r_old + 2 * BH;

  const size_t sofs = (size_t)tstep * BH;
  float* const hout = out + (size_t)layer * TBH + sofs;
  const float a = a_ws[layer * Hn + cn];
  float* const Xl = X + (size_t)layer * BH;
  __bf16* const rnl = r_new + (size_t)layer * BH;

  v8f z = {};
  if (layer == 0) {
    // nx1 = (1-a1)x1 + a1*(stim@w_in + r1@Wr1 + r2@Wfb21)
    v8f accR  = wmma_chain_lds<Hn>(r1 + (size_t)aRow * Hn, sW[0], lane, kb, z);
    accR      = wmma_chain_lds<In>(stimb + (size_t)tstep * Bn * In + (size_t)aRow * In,
                                   sWin, lane, kb, accR);
    v8f accFB = wmma_chain_lds<Hn>(r2 + (size_t)aRow * Hn, sW[1], lane, kb, z);
    store_tile(out + 5 * TBH + sofs, accFB, cm, cn);            // fb21
    update_state(Xl, hout, rnl, a, accR + accFB, cm, cn);
  } else if (layer == 1) {
    // nx2 = (1-a2)x2 + a2*(r1@Wff12 + r2@Wr2 + r3@Wfb32)
    v8f accFF = wmma_chain_lds<Hn>(r1 + (size_t)aRow * Hn, sW[0], lane, kb, z);
    v8f accR  = wmma_chain_lds<Hn>(r2 + (size_t)aRow * Hn, sW[1], lane, kb, z);
    v8f accFB = wmma_chain_lds<Hn>(r3 + (size_t)aRow * Hn, sW[2], lane, kb, z);
    store_tile(out + 3 * TBH + sofs, accFF, cm, cn);            // ff12
    store_tile(out + 6 * TBH + sofs, accFB, cm, cn);            // fb32
    update_state(Xl, hout, rnl, a, accFF + accR + accFB, cm, cn);
  } else {
    // nx3 = (1-a3)x3 + a3*(r2@Wff23 + r3@Wr3)
    v8f accFF = wmma_chain_lds<Hn>(r2 + (size_t)aRow * Hn, sW[0], lane, kb, z);
    v8f accR  = wmma_chain_lds<Hn>(r3 + (size_t)aRow * Hn, sW[1], lane, kb, z);
    store_tile(out + 4 * TBH + sofs, accFF, cm, cn);            // ff23
    update_state(Xl, hout, rnl, a, accFF + accR, cm, cn);
  }
}

// ---------------- prep: Dale-law effective weights -> bf16 ----------------
__global__ void prep_w(const float* __restrict__ wrec1, const float* __restrict__ wrec2,
                       const float* __restrict__ wrec3, const float* __restrict__ wff12,
                       const float* __restrict__ wff23, const float* __restrict__ wfb21,
                       const float* __restrict__ wfb32, const float* __restrict__ m1,
                       const float* __restrict__ m2, const float* __restrict__ m3,
                       __bf16* __restrict__ wdst) {
  int mat = blockIdx.y;
  int idx = blockIdx.x * blockDim.x + threadIdx.x;
  if (idx >= (int)HH) return;
  int k = idx >> 9;  // presynaptic row
  const float* src = wrec1; const float* m = m1; bool sgn = true;
  switch (mat) {
    case 0: src = wrec1; m = m1; sgn = true;  break;
    case 1: src = wrec2; m = m2; sgn = true;  break;
    case 2: src = wrec3; m = m3; sgn = true;  break;
    case 3: src = wff12; m = m1; sgn = false; break;
    case 4: src = wff23; m = m2; sgn = false; break;
    case 5: src = wfb21; m = m2; sgn = false; break;
    case 6: src = wfb32; m = m3; sgn = false; break;
  }
  float w  = fmaxf(src[idx], 0.0f);
  float mk = m[k];
  float eff = sgn ? mk * w : (mk > 0.0f ? w : 0.0f);
  wdst[(size_t)mat * HH + idx] = (__bf16)eff;
}

__global__ void cvt_bf16(const float* __restrict__ src, __bf16* __restrict__ dst, int n) {
  int i = blockIdx.x * blockDim.x + threadIdx.x;
  if (i < n) dst[i] = (__bf16)src[i];
}

__global__ void prep_tau(const float* __restrict__ tg1, const float* __restrict__ tg2,
                         const float* __restrict__ tg3, float* __restrict__ a_ws,
                         float* __restrict__ tau_out) {
  int i = blockIdx.x * blockDim.x + threadIdx.x;
  if (i >= 3 * Hn) return;
  int l = i >> 9, j = i & (Hn - 1);
  const float* tg = (l == 0) ? tg1 : (l == 1) ? tg2 : tg3;
  float s   = 1.0f / (1.0f + expf(-tg[j]));
  float tau = s * (TAU_MAX_C - TAU_MIN_C) + TAU_MIN_C;
  a_ws[i]   = DT_C / tau;
  tau_out[i] = tau;
}

__global__ void init_state(const float* __restrict__ x1, const float* __restrict__ x2,
                           const float* __restrict__ x3, float* __restrict__ X,
                           __bf16* __restrict__ r0) {
  int i = blockIdx.x * blockDim.x + threadIdx.x;
  if (i >= (int)(3 * BH)) return;
  int l = i / (int)BH, j = i % (int)BH;
  const float* x = (l == 0) ? x1 : (l == 1) ? x2 : x3;
  float v = x[j];
  X[i]  = v;
  r0[i] = (__bf16)fmaxf(v, 0.0f);
}

// ---------------- host launch ----------------
extern "C" void kernel_launch(void* const* d_in, const int* in_sizes, int n_in,
                              void* d_out, int out_size, void* d_ws, size_t ws_size,
                              hipStream_t stream) {
  const float* stims = (const float*)d_in[0];
  // d_in[1] = cues (unused by reference)
  const float* w_in  = (const float*)d_in[2];
  const float* m1 = (const float*)d_in[3];
  const float* m2 = (const float*)d_in[4];
  const float* m3 = (const float*)d_in[5];
  const float* wrec1 = (const float*)d_in[6];
  const float* wrec2 = (const float*)d_in[7];
  const float* wrec3 = (const float*)d_in[8];
  const float* wff12 = (const float*)d_in[9];
  const float* wff23 = (const float*)d_in[10];
  const float* wfb21 = (const float*)d_in[11];
  const float* wfb32 = (const float*)d_in[12];
  const float* tg1 = (const float*)d_in[13];
  const float* tg2 = (const float*)d_in[14];
  const float* tg3 = (const float*)d_in[15];
  const float* x1_0 = (const float*)d_in[16];
  const float* x2_0 = (const float*)d_in[17];
  const float* x3_0 = (const float*)d_in[18];

  uint8_t* wsb = (uint8_t*)d_ws;
  __bf16* Weff  = (__bf16*)(wsb + WOFF_WEFF);
  __bf16* winb  = (__bf16*)(wsb + WOFF_WIN);
  __bf16* stimb = (__bf16*)(wsb + WOFF_STIM);
  float*  a_ws  = (float*)(wsb + WOFF_A);
  float*  X     = (float*)(wsb + WOFF_X);
  __bf16* R     = (__bf16*)(wsb + WOFF_R);  // [2][3*BH] ping-pong rates
  float*  out   = (float*)d_out;

  prep_w<<<dim3((unsigned)(HH / 256), 7), 256, 0, stream>>>(
      wrec1, wrec2, wrec3, wff12, wff23, wfb21, wfb32, m1, m2, m3, Weff);
  cvt_bf16<<<(In * Hn + 255) / 256, 256, 0, stream>>>(w_in, winb, In * Hn);
  cvt_bf16<<<(Tn * Bn * In + 255) / 256, 256, 0, stream>>>(stims, stimb, Tn * Bn * In);
  prep_tau<<<(3 * Hn + 255) / 256, 256, 0, stream>>>(tg1, tg2, tg3, a_ws, out + 7 * TBH);
  init_state<<<((int)(3 * BH) + 255) / 256, 256, 0, stream>>>(x1_0, x2_0, x3_0, X, R);

  for (int t = 0; t < Tn; t++) {
    const __bf16* r_old = R + (size_t)(t & 1) * (3 * BH);
    __bf16*       r_new = R + (size_t)((t + 1) & 1) * (3 * BH);
    rnn_step<<<96, 256, 0, stream>>>(Weff, winb, stimb, a_ws, X, r_old, r_new, out, t);
  }
}
